// Transformer_59047210385799
// MI455X (gfx1250) — matrix operand order repaired and checked by reading, not compiled, not run on hardware
//
#include <hip/hip_runtime.h>
#include <math.h>

typedef __attribute__((ext_vector_type(16))) __bf16 v16bf;
typedef __attribute__((ext_vector_type(8)))  float  v8f;

union FragBF { unsigned u[8]; v16bf v; };

__device__ __forceinline__ unsigned short f2bf(float f) {
  unsigned u = __builtin_bit_cast(unsigned, f);
  u += 0x7FFFu + ((u >> 16) & 1u);           // round-to-nearest-even
  return (unsigned short)(u >> 16);
}
__device__ __forceinline__ unsigned pack2bf(float lo, float hi) {
  return (unsigned)f2bf(lo) | ((unsigned)f2bf(hi) << 16);
}

// ---------------------------------------------------------------------------
// C = act(A[M,K] @ W[K,N] + bias)  via bf16 WMMA, fp32 accumulate.
// Block: 256 threads (8 waves), tile 128(M) x 128(N), K-step 32.
// Wave (wr,wc): wr in 0..3 (M), wc in 0..1 (N); wave tile 32x64 = 2x4 wmma.
// Staging: float4 global loads, bf16 packed into LDS (B transposed).
// ---------------------------------------------------------------------------
__global__ __launch_bounds__(256) void gemm_bf16_kernel(
    const float* __restrict__ A, const float* __restrict__ W,
    const float* __restrict__ bias, float* __restrict__ C,
    int M, int N, int K, int act_gelu)
{
  __shared__ unsigned short As[128][34];    // [m][k]  bf16
  __shared__ unsigned short BsT[128][34];   // [n][k]  bf16 (transposed)
  const int tid  = threadIdx.x;
  const int lane = tid & 31;
  const int lh   = lane >> 4;   // half-wave id (0/1)
  const int l15  = lane & 15;
  const int wid  = tid >> 5;
  const int wr   = wid >> 1;    // 0..3  (M)
  const int wc   = wid & 1;     // 0..1  (N)
  const int m0   = blockIdx.y * 128;
  const int n0   = blockIdx.x * 128;

  v8f acc[2][4] = {};

  for (int k0 = 0; k0 < K; k0 += 32) {
    __syncthreads();
#pragma unroll
    for (int j = 0; j < 4; ++j) {              // A tile 128x32, float4 loads
      int i = tid + j * 256;
      int r = i >> 3, c = (i & 7) * 4;
      float4 f = *(const float4*)&A[(size_t)(m0 + r) * K + k0 + c];
      *(unsigned*)&As[r][c]     = pack2bf(f.x, f.y);
      *(unsigned*)&As[r][c + 2] = pack2bf(f.z, f.w);
    }
#pragma unroll
    for (int j = 0; j < 4; ++j) {              // B tile 32x128 -> transposed
      int i = tid + j * 256;
      int r = i >> 5, c = (i & 31) * 4;
      float4 f = *(const float4*)&W[(size_t)(k0 + r) * N + n0 + c];
      BsT[c    ][r] = f2bf(f.x);
      BsT[c + 1][r] = f2bf(f.y);
      BsT[c + 2][r] = f2bf(f.z);
      BsT[c + 3][r] = f2bf(f.w);
    }
    __syncthreads();

    FragBF bfrag[4];
#pragma unroll
    for (int ni = 0; ni < 4; ++ni) {           // B frag: col=lane, K pairs across vgprs
      int col = wc * 64 + ni * 16 + l15;
#pragma unroll
      for (int j = 0; j < 8; ++j)
        bfrag[ni].u[j] = *(const unsigned*)&BsT[col][(j >> 2) * 16 + lh * 8 + 2 * (j & 3)];
    }
#pragma unroll
    for (int mi = 0; mi < 2; ++mi) {
      FragBF afrag;                            // A frag: row=lane, half-split K pairs
      int mrow = wr * 32 + mi * 16 + l15;
      int kb0 = lh * 8, kb1 = 16 + lh * 8;
#pragma unroll
      for (int i = 0; i < 4; ++i) {
        afrag.u[i]     = *(const unsigned*)&As[mrow][kb0 + 2 * i];
        afrag.u[4 + i] = *(const unsigned*)&As[mrow][kb1 + 2 * i];
      }
#pragma unroll
      for (int ni = 0; ni < 4; ++ni)
        acc[mi][ni] = __builtin_amdgcn_wmma_f32_16x16x32_bf16(
            false, afrag.v, false, bfrag[ni].v, (short)0, acc[mi][ni], false, false);
    }
  }

#pragma unroll
  for (int mi = 0; mi < 2; ++mi)
#pragma unroll
    for (int ni = 0; ni < 4; ++ni) {
      int col = n0 + wc * 64 + ni * 16 + l15;
      float bv = bias[col];
#pragma unroll
      for (int r = 0; r < 8; ++r) {            // C layout: row = r + 8*half
        int row = m0 + wr * 32 + mi * 16 + r + 8 * lh;
        float x = acc[mi][ni][r] + bv;
        if (act_gelu)
          x = 0.5f * x * (1.0f + erff(x * 0.70710678118654752f));
        C[(size_t)row * N + col] = x;
      }
    }
}

// ---------------------------------------------------------------------------
// Flash attention, causal, NH=16, D=64. Block=128 threads (4 waves), one
// (batch, head, 64-row q tile) per block; each wave owns 16 q rows.
// ---------------------------------------------------------------------------
__global__ __launch_bounds__(128) void flash_attn_kernel(
    const float* __restrict__ qg, const float* __restrict__ kg,
    const float* __restrict__ vg, float* __restrict__ og)
{
  const int Tt = 1024, Hh = 1024;
  __shared__ unsigned short Ks[32][66];    // [kpos][d]  bf16
  __shared__ unsigned short VsT[64][34];   // [d][kpos]  bf16 (transposed)
  __shared__ unsigned short Ps[4][16][34]; // per-wave P [qrow][kpos] bf16

  const int tid  = threadIdx.x;
  const int lane = tid & 31;
  const int lh   = lane >> 4;
  const int l15  = lane & 15;
  const int w    = tid >> 5;         // wave 0..3
  const int qt   = blockIdx.x;       // q tile of 64
  const int head = blockIdx.y;
  const int b    = blockIdx.z;
  const float scale = 0.125f;        // 1/sqrt(D=64), folded into Q

  FragBF qf[2];                      // Q 16x64 = two 16x32 A-frags
  {
    const float* qp = qg + ((size_t)(b * Tt + qt * 64 + w * 16 + l15)) * Hh + head * 64;
#pragma unroll
    for (int f = 0; f < 2; ++f) {
      int b0 = f * 32 + lh * 8;
      int b1 = f * 32 + 16 + lh * 8;
#pragma unroll
      for (int i = 0; i < 4; ++i) {
        qf[f].u[i]     = pack2bf(qp[b0 + 2 * i] * scale, qp[b0 + 2 * i + 1] * scale);
        qf[f].u[4 + i] = pack2bf(qp[b1 + 2 * i] * scale, qp[b1 + 2 * i + 1] * scale);
      }
    }
  }

  v8f o[4] = {};
  float m[8], l[8];
#pragma unroll
  for (int r = 0; r < 8; ++r) { m[r] = -INFINITY; l[r] = 0.f; }

  const int qbase = qt * 64 + w * 16;
  const int kend  = (qt + 1) * 64;

  for (int kt = 0; kt < kend; kt += 32) {
    __syncthreads();
#pragma unroll
    for (int j = 0; j < 4; ++j) {              // stage K,V 32x64 tiles, float4 loads
      int i = tid + j * 128;
      int r = i >> 4, c = (i & 15) * 4;
      size_t gidx = ((size_t)(b * Tt + kt + r)) * Hh + head * 64 + c;
      float4 fk = *(const float4*)&kg[gidx];
      float4 fv = *(const float4*)&vg[gidx];
      *(unsigned*)&Ks[r][c]     = pack2bf(fk.x, fk.y);
      *(unsigned*)&Ks[r][c + 2] = pack2bf(fk.z, fk.w);
      VsT[c    ][r] = f2bf(fv.x);
      VsT[c + 1][r] = f2bf(fv.y);
      VsT[c + 2][r] = f2bf(fv.z);
      VsT[c + 3][r] = f2bf(fv.w);
    }
    __syncthreads();

    float sv[2][8];
#pragma unroll
    for (int tn = 0; tn < 2; ++tn) {           // S = Q @ K^T, 16x16 tiles
      v8f s = {};
#pragma unroll
      for (int f = 0; f < 2; ++f) {
        FragBF kf;                             // B frag of K^T: col=kpos, K-dim=d
        int kr = tn * 16 + l15;
#pragma unroll
        for (int j = 0; j < 8; ++j)
          kf.u[j] = *(const unsigned*)&Ks[kr][f * 32 + (j >> 2) * 16 + lh * 8 + 2 * (j & 3)];
        s = __builtin_amdgcn_wmma_f32_16x16x32_bf16(
            false, qf[f].v, false, kf.v, (short)0, s, false, false);
      }
      int kcol = kt + tn * 16 + l15;
#pragma unroll
      for (int r = 0; r < 8; ++r) {
        int qr = qbase + r + 8 * lh;
        sv[tn][r] = (kcol <= qr) ? s[r] : -INFINITY;   // causal mask
      }
    }

    float pvv[2][8];
#pragma unroll
    for (int r = 0; r < 8; ++r) {              // online softmax per row
      float rm = fmaxf(sv[0][r], sv[1][r]);
#pragma unroll
      for (int off = 8; off >= 1; off >>= 1)
        rm = fmaxf(rm, __shfl_xor(rm, off, 16));
      float mnew = fmaxf(m[r], rm);
      float corr = __expf(m[r] - mnew);
      float p0 = __expf(sv[0][r] - mnew);
      float p1 = __expf(sv[1][r] - mnew);
      pvv[0][r] = p0; pvv[1][r] = p1;
      float rs = p0 + p1;
#pragma unroll
      for (int off = 8; off >= 1; off >>= 1)
        rs += __shfl_xor(rs, off, 16);
      l[r] = l[r] * corr + rs;
      m[r] = mnew;
#pragma unroll
      for (int dt = 0; dt < 4; ++dt)
        o[dt][r] *= corr;
    }

#pragma unroll
    for (int tn = 0; tn < 2; ++tn)             // P -> LDS (C-layout to A-layout fixup)
#pragma unroll
      for (int r = 0; r < 8; ++r)
        Ps[w][r + 8 * lh][tn * 16 + l15] = f2bf(pvv[tn][r]);

    asm volatile("s_wait_dscnt 0" ::: "memory");  // wave-local LDS ordering

    FragBF pf;                                  // P as A-frag 16x32
    {
      int kb0 = lh * 8, kb1 = 16 + lh * 8;
#pragma unroll
      for (int i = 0; i < 4; ++i) {
        pf.u[i]     = *(const unsigned*)&Ps[w][l15][kb0 + 2 * i];
        pf.u[4 + i] = *(const unsigned*)&Ps[w][l15][kb1 + 2 * i];
      }
    }
#pragma unroll
    for (int dt = 0; dt < 4; ++dt) {            // O += P @ V
      FragBF vf;
      int cold = dt * 16 + l15;
#pragma unroll
      for (int j = 0; j < 8; ++j)
        vf.u[j] = *(const unsigned*)&VsT[cold][(j >> 2) * 16 + lh * 8 + 2 * (j & 3)];
      o[dt] = __builtin_amdgcn_wmma_f32_16x16x32_bf16(
          false, pf.v, false, vf.v, (short)0, o[dt], false, false);
    }
  }

#pragma unroll
  for (int dt = 0; dt < 4; ++dt)
#pragma unroll
    for (int r = 0; r < 8; ++r) {
      int qrow = qbase + r + 8 * lh;
      og[((size_t)(b * Tt + qrow)) * Hh + head * 64 + dt * 16 + l15] = o[dt][r] / l[r];
    }
}

// ---------------------------------------------------------------------------
// out = LayerNorm(a + add) * g + be     (add may be null). One row per block.
// ---------------------------------------------------------------------------
__global__ __launch_bounds__(256) void ln_kernel(
    const float* __restrict__ a, const float* __restrict__ addp,
    const float* __restrict__ g, const float* __restrict__ be,
    float* __restrict__ out)
{
  const int Hh = 1024;
  __shared__ float red[8];
  int row = blockIdx.x, tid = threadIdx.x;
  int lane = tid & 31, wid = tid >> 5;
  float v[4], s = 0.f;
#pragma unroll
  for (int j = 0; j < 4; ++j) {
    int c = tid + j * 256;
    float x = a[(size_t)row * Hh + c];
    if (addp) x += addp[(size_t)row * Hh + c];
    v[j] = x; s += x;
  }
#pragma unroll
  for (int off = 16; off >= 1; off >>= 1) s += __shfl_xor(s, off, 32);
  if (lane == 0) red[wid] = s;
  __syncthreads();
  float tot = 0.f;
#pragma unroll
  for (int i = 0; i < 8; ++i) tot += red[i];
  float mu = tot * (1.0f / 1024.0f);
  __syncthreads();
  float var = 0.f;
#pragma unroll
  for (int j = 0; j < 4; ++j) { float d = v[j] - mu; var += d * d; }
#pragma unroll
  for (int off = 16; off >= 1; off >>= 1) var += __shfl_xor(var, off, 32);
  if (lane == 0) red[wid] = var;
  __syncthreads();
  float vtot = 0.f;
#pragma unroll
  for (int i = 0; i < 8; ++i) vtot += red[i];
  float inv = rsqrtf(vtot * (1.0f / 1024.0f) + 1e-5f);
#pragma unroll
  for (int j = 0; j < 4; ++j) {
    int c = tid + j * 256;
    out[(size_t)row * Hh + c] = (v[j] - mu) * inv * g[c] + be[c];
  }
}

// ---------------------------------------------------------------------------
// h[bt] = tok_emb[x[bt]] + pos_emb[bt % T]
// ---------------------------------------------------------------------------
__global__ __launch_bounds__(256) void embed_kernel(
    const int* __restrict__ x, const float* __restrict__ tok,
    const float* __restrict__ pos, float* __restrict__ h)
{
  int bt = blockIdx.x;
  int t = bt & 1023;
  int token = x[bt];
#pragma unroll
  for (int j = 0; j < 4; ++j) {
    int c = threadIdx.x + j * 256;
    h[(size_t)bt * 1024 + c] = tok[(size_t)token * 1024 + c] + pos[(size_t)t * 1024 + c];
  }
}

// ---------------------------------------------------------------------------
extern "C" void kernel_launch(void* const* d_in, const int* in_sizes, int n_in,
                              void* d_out, int out_size, void* d_ws, size_t ws_size,
                              hipStream_t stream)
{
  (void)in_sizes; (void)n_in; (void)out_size; (void)ws_size;
  const int B = 2, T = 1024, H = 1024, NH = 16, L = 6, V = 8192, FF = 4096;
  const int M = B * T;

  const int*   x     = (const int*)  d_in[0];
  const float* tok   = (const float*)d_in[1];
  const float* pos   = (const float*)d_in[2];
  const float* Wq    = (const float*)d_in[3];
  const float* bq    = (const float*)d_in[4];
  const float* Wk    = (const float*)d_in[5];
  const float* bk    = (const float*)d_in[6];
  const float* Wv    = (const float*)d_in[7];
  const float* bv    = (const float*)d_in[8];
  const float* Wo    = (const float*)d_in[9];
  const float* bo    = (const float*)d_in[10];
  const float* W1    = (const float*)d_in[11];
  const float* b1    = (const float*)d_in[12];
  const float* W2    = (const float*)d_in[13];
  const float* b2    = (const float*)d_in[14];
  const float* ln1g  = (const float*)d_in[15];
  const float* ln1b  = (const float*)d_in[16];
  const float* ln2g  = (const float*)d_in[17];
  const float* ln2b  = (const float*)d_in[18];
  const float* lnfg  = (const float*)d_in[19];
  const float* lnfb  = (const float*)d_in[20];
  const float* headW = (const float*)d_in[21];
  const float* headb = (const float*)d_in[22];

  float* ws   = (float*)d_ws;
  float* h    = ws;                       // [M,H]
  float* q    = h    + (size_t)M * H;     // [M,H]
  float* kbuf = q    + (size_t)M * H;     // [M,H]
  float* vbuf = kbuf + (size_t)M * H;     // [M,H]
  float* attn = vbuf + (size_t)M * H;     // [M,H]
  float* tmp  = attn + (size_t)M * H;     // [M,H]
  float* ff   = tmp  + (size_t)M * H;     // [M,FF]

  embed_kernel<<<M, 256, 0, stream>>>(x, tok, pos, h);

  dim3 gH(H / 128, M / 128), gFF(FF / 128, M / 128), gV(V / 128, M / 128);
  for (int l = 0; l < L; ++l) {
    size_t wHH = (size_t)l * H * H, wb = (size_t)l * H;
    gemm_bf16_kernel<<<gH, 256, 0, stream>>>(h, Wq + wHH, bq + wb, q,    M, H, H, 0);
    gemm_bf16_kernel<<<gH, 256, 0, stream>>>(h, Wk + wHH, bk + wb, kbuf, M, H, H, 0);
    gemm_bf16_kernel<<<gH, 256, 0, stream>>>(h, Wv + wHH, bv + wb, vbuf, M, H, H, 0);
    flash_attn_kernel<<<dim3(T / 64, NH, B), 128, 0, stream>>>(q, kbuf, vbuf, attn);
    gemm_bf16_kernel<<<gH, 256, 0, stream>>>(attn, Wo + wHH, bo + wb, tmp, M, H, H, 0);
    ln_kernel<<<M, 256, 0, stream>>>(h, tmp, ln1g + wb, ln1b + wb, h);
    gemm_bf16_kernel<<<gFF, 256, 0, stream>>>(h, W1 + (size_t)l * H * FF, b1 + (size_t)l * FF, ff, M, FF, H, 1);
    gemm_bf16_kernel<<<gH, 256, 0, stream>>>(ff, W2 + (size_t)l * FF * H, b2 + wb, tmp, M, H, FF, 0);
    ln_kernel<<<M, 256, 0, stream>>>(h, tmp, ln2g + wb, ln2b + wb, h);
  }
  ln_kernel<<<M, 256, 0, stream>>>(h, nullptr, lnfg, lnfb, h);
  gemm_bf16_kernel<<<gV, 256, 0, stream>>>(h, headW, headb, (float*)d_out, M, V, H, 0);
}
